// PatchLocalPoolPointnet_34626026340942
// MI455X (gfx1250) — matrix operand order repaired and checked by reading, not compiled, not run on hardware
//
#include <hip/hip_runtime.h>
#include <float.h>

typedef unsigned short u16;
typedef __attribute__((ext_vector_type(16))) __bf16        v16bf;
typedef __attribute__((ext_vector_type(8)))  float         v8f;
typedef __attribute__((ext_vector_type(4)))  unsigned int  v4u;

#define HID   128
#define KX2   256
#define BB    8
#define TT    30000
#define NPT   240000
#define R2    4096
#define NBLK  5
#define PTS_PER_BLK 64
#define THREADS 128

__device__ __forceinline__ u16 f2bf(float f) {
    unsigned int u = __float_as_uint(f);
    u += 0x7FFFu + ((u >> 16) & 1u);   // round-to-nearest-even
    return (u16)(u >> 16);
}

union FragBF { v16bf v; v4u q[2]; unsigned int d[8]; };

// A fragment (16x32 bf16): lane's 16 elems are K {koff..koff+7, koff+16..koff+23}
__device__ __forceinline__ v16bf load_a(const u16* p) {
    FragBF f;
    f.q[0] = *(const v4u*)(p);
    f.q[1] = *(const v4u*)(p + 16);
    return f.v;
}
__device__ __forceinline__ v16bf load_a_relu(const u16* p) {
    FragBF f;
    f.q[0] = *(const v4u*)(p);
    f.q[1] = *(const v4u*)(p + 16);
#pragma unroll
    for (int i = 0; i < 8; ++i) {
        unsigned int v = f.d[i];
        unsigned int m = (((v >> 15) & 1u) ? 0u : 0x0000FFFFu) |
                         (((v >> 31) & 1u) ? 0u : 0xFFFF0000u);
        f.d[i] = v & m;                 // bf16 relu: zero negative halves
    }
    return f.v;
}
// B fragment (32x16 bf16): lane's 16 elems are K {koff..koff+7, koff+16..koff+23} of one column
__device__ __forceinline__ v16bf load_b(const u16* p) {
    FragBF f;
    f.q[0] = *(const v4u*)(p);
    f.q[1] = *(const v4u*)(p + 16);
    return f.v;
}

// MODE 0: fc_pos + resnet block 0   (x staged from points)
// MODE 1: resnet block i>=1         (x = [net | sum of gathered bin maxima])
// MODE 2: c = net @ fc_c + b, fused scatter-add into output planes
template <int MODE>
__global__ __launch_bounds__(THREADS) void gemm_block_kernel(
    const float* __restrict__ points,
    const float* __restrict__ fc_pos_w,
    const float* __restrict__ fc_pos_b,
    float* __restrict__ net,
    const float* __restrict__ bins,
    const int* __restrict__ idx_xz,
    const int* __restrict__ idx_xy,
    const int* __restrict__ idx_yz,
    const u16* __restrict__ W0t, const float* __restrict__ b0,
    const u16* __restrict__ W1t, const float* __restrict__ b1,
    const u16* __restrict__ Wst,
    float* __restrict__ outp)
{
    __shared__ u16   xtile[PTS_PER_BLK * KX2];              // 32 KB
    __shared__ u16   htile[(THREADS / 32) * 16 * HID];      // 16 KB
    __shared__ float pts[PTS_PER_BLK * 3];

    const int t  = threadIdx.x;
    const int g0 = blockIdx.x * PTS_PER_BLK;

    // ---------------- stage activation tile (bf16) ----------------
    if (MODE == 0) {
        for (int i = t; i < PTS_PER_BLK * 3; i += THREADS)
            pts[i] = points[(size_t)g0 * 3 + i];
        __syncthreads();
#pragma unroll 2
        for (int j = 0; j < (PTS_PER_BLK * KX2) / THREADS; ++j) {
            int e = j * THREADS + t;
            int pt = e >> 8, k = e & 255;
            float v = fc_pos_b[k]
                    + pts[pt * 3 + 0] * fc_pos_w[k]
                    + pts[pt * 3 + 1] * fc_pos_w[256 + k]
                    + pts[pt * 3 + 2] * fc_pos_w[512 + k];
            xtile[pt * KX2 + k] = f2bf(v);
        }
    } else if (MODE == 1) {
#pragma unroll 2
        for (int j = 0; j < (PTS_PER_BLK * KX2) / THREADS; ++j) {
            int e = j * THREADS + t;
            int pt = e >> 8, k = e & 255;
            int g = g0 + pt;
            float v;
            if (k < HID) {
                v = net[(size_t)g * HID + k];
            } else {
                int kk = k - HID;
                int b  = g / TT;
                const float* bb = bins + (size_t)b * 3 * R2 * HID + kk;
                v = bb[(size_t)(0 * R2 + idx_xz[g]) * HID]
                  + bb[(size_t)(1 * R2 + idx_xy[g]) * HID]
                  + bb[(size_t)(2 * R2 + idx_yz[g]) * HID];
            }
            xtile[pt * KX2 + k] = f2bf(v);
        }
    } else {
#pragma unroll 2
        for (int j = 0; j < (PTS_PER_BLK * HID) / THREADS; ++j) {
            int e = j * THREADS + t;
            int pt = e >> 7, k = e & 127;
            xtile[pt * HID + k] = f2bf(net[(size_t)(g0 + pt) * HID + k]);
        }
    }
    __syncthreads();

    const int lane = t & 31;
    const int w    = t >> 5;
    const int mrow = lane & 15;
    const int hi   = lane >> 4;
    const int rowA = w * 16 + mrow;

    v8f acc[8];

    if (MODE != 2) {
        // ---- GEMM1: h = relu(x) @ W0 + b0 ----
#pragma unroll
        for (int nt = 0; nt < 8; ++nt) {
            float bv = b0[nt * 16 + mrow];
            v8f a = {bv, bv, bv, bv, bv, bv, bv, bv};
            acc[nt] = a;
        }
#pragma unroll
        for (int kc = 0; kc < KX2 / 32; ++kc) {
            const int kb = kc * 32;
            v16bf a = load_a_relu(&xtile[rowA * KX2 + kb + 8 * hi]);
#pragma unroll
            for (int nt = 0; nt < 8; ++nt) {
                v16bf bm = load_b(W0t + (size_t)(nt * 16 + mrow) * KX2 + kb + 8 * hi);
                acc[nt] = __builtin_amdgcn_wmma_f32_16x16x32_bf16(
                    false, a, false, bm, (short)0, acc[nt], false, false);
            }
        }
        // relu(h) -> per-wave LDS tile (bf16) for re-fragmentation
        u16* hh = &htile[w * 16 * HID];
#pragma unroll
        for (int nt = 0; nt < 8; ++nt) {
            const int n = nt * 16 + mrow;
#pragma unroll
            for (int r = 0; r < 8; ++r) {
                const int m = r + 8 * hi;
                float v = acc[nt][r];
                v = v > 0.f ? v : 0.f;
                hh[m * HID + n] = f2bf(v);
            }
        }
        __syncthreads();

        // ---- out = x @ Ws + relu(h) @ W1 + b1 ----
#pragma unroll
        for (int nt = 0; nt < 8; ++nt) {
            float bv = b1[nt * 16 + mrow];
            v8f a = {bv, bv, bv, bv, bv, bv, bv, bv};
            acc[nt] = a;
        }
#pragma unroll
        for (int kc = 0; kc < KX2 / 32; ++kc) {
            const int kb = kc * 32;
            v16bf a = load_a(&xtile[rowA * KX2 + kb + 8 * hi]);   // raw x (shortcut)
#pragma unroll
            for (int nt = 0; nt < 8; ++nt) {
                v16bf bm = load_b(Wst + (size_t)(nt * 16 + mrow) * KX2 + kb + 8 * hi);
                acc[nt] = __builtin_amdgcn_wmma_f32_16x16x32_bf16(
                    false, a, false, bm, (short)0, acc[nt], false, false);
            }
        }
#pragma unroll
        for (int kc = 0; kc < HID / 32; ++kc) {
            const int kb = kc * 32;
            v16bf a = load_a(&hh[mrow * HID + kb + 8 * hi]);      // relu(h)
#pragma unroll
            for (int nt = 0; nt < 8; ++nt) {
                v16bf bm = load_b(W1t + (size_t)(nt * 16 + mrow) * HID + kb + 8 * hi);
                acc[nt] = __builtin_amdgcn_wmma_f32_16x16x32_bf16(
                    false, a, false, bm, (short)0, acc[nt], false, false);
            }
        }
        // store net (f32)
#pragma unroll
        for (int nt = 0; nt < 8; ++nt) {
            const int n = nt * 16 + mrow;
#pragma unroll
            for (int r = 0; r < 8; ++r) {
                const int g = g0 + w * 16 + r + 8 * hi;
                net[(size_t)g * HID + n] = acc[nt][r];
            }
        }
    } else {
        // ---- c = net @ fc_c + b, fused plane scatter-add ----
#pragma unroll
        for (int nt = 0; nt < 8; ++nt) {
            float bv = b0[nt * 16 + mrow];
            v8f a = {bv, bv, bv, bv, bv, bv, bv, bv};
            acc[nt] = a;
        }
#pragma unroll
        for (int kc = 0; kc < HID / 32; ++kc) {
            const int kb = kc * 32;
            v16bf a = load_a(&xtile[rowA * HID + kb + 8 * hi]);
#pragma unroll
            for (int nt = 0; nt < 8; ++nt) {
                v16bf bm = load_b(W0t + (size_t)(nt * 16 + mrow) * HID + kb + 8 * hi);
                acc[nt] = __builtin_amdgcn_wmma_f32_16x16x32_bf16(
                    false, a, false, bm, (short)0, acc[nt], false, false);
            }
        }
        int bn0[8], bn1[8], bn2[8], bof[8];
#pragma unroll
        for (int r = 0; r < 8; ++r) {
            const int g = g0 + w * 16 + r + 8 * hi;
            bn0[r] = idx_xz[g]; bn1[r] = idx_xy[g]; bn2[r] = idx_yz[g];
            bof[r] = g / TT;
        }
#pragma unroll
        for (int nt = 0; nt < 8; ++nt) {
            const int ch = nt * 16 + mrow;
#pragma unroll
            for (int r = 0; r < 8; ++r) {
                const float v = acc[nt][r];
                const int b = bof[r];
                atomicAdd(&outp[((size_t)((0 * BB + b) * HID + ch)) * R2 + bn0[r]], v);
                atomicAdd(&outp[((size_t)((1 * BB + b) * HID + ch)) * R2 + bn1[r]], v);
                atomicAdd(&outp[((size_t)((2 * BB + b) * HID + ch)) * R2 + bn2[r]], v);
            }
        }
    }
}

__global__ void fill_kernel(float* __restrict__ p, float v, size_t n) {
    size_t i = (size_t)blockIdx.x * blockDim.x + threadIdx.x;
    if (i < n) p[i] = v;
}

// dst[(ib*N + n)*K + k] = bf16(src[(ib*K + k)*N + n])
__global__ void transpose_bf16_kernel(const float* __restrict__ src, u16* __restrict__ dst,
                                      int nb, int K, int N) {
    long long i = (long long)blockIdx.x * blockDim.x + threadIdx.x;
    long long total = (long long)nb * K * N;
    if (i >= total) return;
    int kk = (int)(i % K);
    long long r = i / K;
    int n  = (int)(r % N);
    int ib = (int)(r / N);
    dst[i] = f2bf(src[((long long)ib * K + kk) * N + n]);
}

__global__ void segmax_kernel(const float* __restrict__ net,
                              const int* __restrict__ i0, const int* __restrict__ i1,
                              const int* __restrict__ i2, float* __restrict__ bins) {
    size_t i = (size_t)blockIdx.x * blockDim.x + threadIdx.x;
    if (i >= (size_t)NPT * HID) return;
    int g  = (int)(i >> 7);
    int ch = (int)(i & 127);
    float v = net[i];
    int b = g / TT;
    float* base = bins + (size_t)b * 3 * R2 * HID + ch;
    int bn[3] = {i0[g], i1[g], i2[g]};
#pragma unroll
    for (int p = 0; p < 3; ++p) {
        float* addr = base + (size_t)(p * R2 + bn[p]) * HID;
        if (v >= 0.f) atomicMax((int*)addr, __float_as_int(v));
        else          atomicMin((unsigned int*)addr, __float_as_uint(v));
    }
}

__global__ void counts_kernel(const int* __restrict__ i0, const int* __restrict__ i1,
                              const int* __restrict__ i2, float* __restrict__ counts) {
    int g = blockIdx.x * blockDim.x + threadIdx.x;
    if (g >= NPT) return;
    int b = g / TT;
    atomicAdd(&counts[(0 * BB + b) * R2 + i0[g]], 1.f);
    atomicAdd(&counts[(1 * BB + b) * R2 + i1[g]], 1.f);
    atomicAdd(&counts[(2 * BB + b) * R2 + i2[g]], 1.f);
}

__global__ void finalize_kernel(float* __restrict__ out, const float* __restrict__ counts, size_t n) {
    size_t i = (size_t)blockIdx.x * blockDim.x + threadIdx.x;
    if (i >= n) return;
    size_t pb = i >> 19;                 // / (HID*R2)
    int bin = (int)(i & (R2 - 1));
    float c = counts[pb * R2 + bin];
    out[i] = out[i] / fmaxf(c, 1.f);
}

extern "C" void kernel_launch(void* const* d_in, const int* in_sizes, int n_in,
                              void* d_out, int out_size, void* d_ws, size_t ws_size,
                              hipStream_t stream) {
    const float* points   = (const float*)d_in[0];
    const int*   idx_xz   = (const int*)d_in[1];
    const int*   idx_xy   = (const int*)d_in[2];
    const int*   idx_yz   = (const int*)d_in[3];
    const float* fc_pos_w = (const float*)d_in[4];
    const float* fc_pos_b = (const float*)d_in[5];
    const float* fc0_w    = (const float*)d_in[6];
    const float* fc0_b    = (const float*)d_in[7];
    const float* fc1_w    = (const float*)d_in[8];
    const float* fc1_b    = (const float*)d_in[9];
    const float* sc_w     = (const float*)d_in[10];
    const float* fc_c_w   = (const float*)d_in[11];
    const float* fc_c_b   = (const float*)d_in[12];
    float* out = (float*)d_out;

    char* ws = (char*)d_ws;
    constexpr size_t OFF_NET  = 0;                                   // 240000*128*4
    constexpr size_t OFF_BINS = 122880000;                           // 8*3*4096*128*4
    constexpr size_t OFF_CNT  = OFF_BINS + 50331648;                 // 3*8*4096*4
    constexpr size_t OFF_W0T  = OFF_CNT + 393216;
    constexpr size_t OFF_W1T  = OFF_W0T + (size_t)5 * 128 * 256 * 2;
    constexpr size_t OFF_WST  = OFF_W1T + (size_t)5 * 128 * 128 * 2;
    constexpr size_t OFF_FCT  = OFF_WST + (size_t)5 * 128 * 256 * 2;

    float* net    = (float*)(ws + OFF_NET);
    float* bins   = (float*)(ws + OFF_BINS);
    float* counts = (float*)(ws + OFF_CNT);
    u16*   W0t    = (u16*)(ws + OFF_W0T);
    u16*   W1t    = (u16*)(ws + OFF_W1T);
    u16*   Wst    = (u16*)(ws + OFF_WST);
    u16*   Fct    = (u16*)(ws + OFF_FCT);

    const size_t outN = (size_t)3 * BB * HID * R2;
    fill_kernel<<<(int)((outN + 255) / 256), 256, 0, stream>>>(out, 0.f, outN);
    fill_kernel<<<(3 * BB * R2 + 255) / 256, 256, 0, stream>>>(counts, 0.f, (size_t)3 * BB * R2);

    transpose_bf16_kernel<<<(5 * 256 * 128 + 255) / 256, 256, 0, stream>>>(fc0_w, W0t, NBLK, 256, 128);
    transpose_bf16_kernel<<<(5 * 128 * 128 + 255) / 256, 256, 0, stream>>>(fc1_w, W1t, NBLK, 128, 128);
    transpose_bf16_kernel<<<(5 * 256 * 128 + 255) / 256, 256, 0, stream>>>(sc_w,  Wst, NBLK, 256, 128);
    transpose_bf16_kernel<<<(128 * 128 + 255) / 256, 256, 0, stream>>>(fc_c_w, Fct, 1, 128, 128);

    const int nGemm = NPT / PTS_PER_BLK;   // 3750, exact

    gemm_block_kernel<0><<<nGemm, THREADS, 0, stream>>>(
        points, fc_pos_w, fc_pos_b, net, nullptr, idx_xz, idx_xy, idx_yz,
        W0t, fc0_b, W1t, fc1_b, Wst, nullptr);

    const size_t binsN = (size_t)BB * 3 * R2 * HID;
    for (int i = 1; i < NBLK; ++i) {
        fill_kernel<<<(int)((binsN + 255) / 256), 256, 0, stream>>>(bins, -FLT_MAX, binsN);
        segmax_kernel<<<(int)(((size_t)NPT * HID + 255) / 256), 256, 0, stream>>>(
            net, idx_xz, idx_xy, idx_yz, bins);
        gemm_block_kernel<1><<<nGemm, THREADS, 0, stream>>>(
            nullptr, nullptr, nullptr, net, bins, idx_xz, idx_xy, idx_yz,
            W0t + (size_t)i * 128 * 256, fc0_b + i * 128,
            W1t + (size_t)i * 128 * 128, fc1_b + i * 128,
            Wst + (size_t)i * 128 * 256, nullptr);
    }

    gemm_block_kernel<2><<<nGemm, THREADS, 0, stream>>>(
        nullptr, nullptr, nullptr, net, nullptr, idx_xz, idx_xy, idx_yz,
        Fct, fc_c_b, nullptr, nullptr, nullptr, out);

    counts_kernel<<<(NPT + 255) / 256, 256, 0, stream>>>(idx_xz, idx_xy, idx_yz, counts);
    finalize_kernel<<<(int)((outN + 255) / 256), 256, 0, stream>>>(out, counts, outN);

    (void)in_sizes; (void)n_in; (void)out_size; (void)ws_size;
}
